// Net_86689619902982
// MI455X (gfx1250) — compile-verified, matrix operand order checked by reading, not run on hardware
//
#include <hip/hip_runtime.h>
#include <math.h>

// ---------------------------------------------------------------------------
// Types for CDNA5 WMMA (wave32, 16x16x32 bf16 -> f32)
// ---------------------------------------------------------------------------
typedef __attribute__((ext_vector_type(16))) __bf16 v16bf;
typedef __attribute__((ext_vector_type(8)))  float  v8f;

union FragU {
    unsigned u[8];
    v16bf    v;
};

__device__ __forceinline__ unsigned short f2bf(float f) {
    union { float f; unsigned u; } x; x.f = f;
    unsigned u = x.u;
    u += 0x7FFFu + ((u >> 16) & 1u);   // round-to-nearest-even
    return (unsigned short)(u >> 16);
}
__device__ __forceinline__ float bf2f(unsigned short h) {
    union { unsigned u; float f; } x; x.u = ((unsigned)h) << 16;
    return x.f;
}

// ---------------------------------------------------------------------------
// Implicit-GEMM WMMA kernel.  C[M,N] = act(A[M,K] * W[K,N] + bias)
//
// A (bf16):
//   amode 0: dense row-major [M,K], K%32==0  (vector b128 staging)
//   amode 1: generic im2col gather from NHWC activation (any C/K)
//   amode 2: channel-aligned im2col, requires C%32==0 (vector b128 staging,
//            no division in the K loop)
// W (f32, converted to bf16 while staging):
//   bmode 0: FC layout [N,K], K%32==0            (float4-friendly)
//   bmode 1: generic conv layout [N,C,KH,KW]
//   bmode 2: channel-aligned conv layout, C%32==0 (affine stride KH*KW)
//
// im2col k index: k = (kh*KW + kw)*C + c  (c fastest; matches NHWC)
//
// Block: 256 threads = 8 waves. Block tile 128x128. Wave tile 32(M)x64(N):
// 2 A-frags x 4 B-frags -> 8 v_wmma_f32_16x16x32_bf16 per 32-wide K step.
// ---------------------------------------------------------------------------
#define BT   128
#define LDSR 40   // LDS row stride in bf16 elems: 80B rows -> 16B-aligned bases

__global__ void __launch_bounds__(256)
wmma_gemm_k(const unsigned short* __restrict__ A,
            const float* __restrict__ W,
            const float* __restrict__ bias,
            unsigned short* __restrict__ outB,
            float* __restrict__ outF,
            int M, int N, int K,
            int amode, int H, int Wd, int C, int OH, int OW,
            int KH, int KW, int stride, int pad,
            int bmode, int relu, int outf32)
{
    __shared__ unsigned short As[BT][LDSR];
    __shared__ unsigned short Bs[BT][LDSR];

    const int tid  = threadIdx.x;
    const int lane = tid & 31;
    const int wave = tid >> 5;        // 0..7
    const int wrow = wave >> 1;       // 0..3 -> M offset 32*wrow
    const int wcol = wave & 1;        // 0..1 -> N offset 64*wcol
    const int hi   = lane >> 4;       // lane half (K half / M+8 select)
    const int lr   = lane & 15;

    const int blockM = blockIdx.y * BT;
    const int blockN = blockIdx.x * BT;

    // staging assignment: each thread owns one row and a 16-elem K half
    const int srow = tid >> 1;          // 0..127
    const int klo  = (tid & 1) * 16;    // 0 or 16

    // hoisted per-row info for A
    const int  am    = blockM + srow;
    const bool am_ok = (am < M);
    int ab = 0, aihb = 0, aiwb = 0;
    if (amode != 0) {
        const int ohow = OH * OW;
        ab      = am / ohow;
        int rem = am - ab * ohow;
        int aoh = rem / OW;
        int aow = rem - aoh * OW;
        aihb = aoh * stride - pad;
        aiwb = aow * stride - pad;
    }
    const int  bn    = blockN + srow;
    const bool bn_ok = (bn < N);

    v8f zero = {0.f, 0.f, 0.f, 0.f, 0.f, 0.f, 0.f, 0.f};
    v8f acc[2][4];
#pragma unroll
    for (int i = 0; i < 2; i++)
#pragma unroll
        for (int j = 0; j < 4; j++) acc[i][j] = zero;

    // incremental decomposition of k0 -> (cbase, kw, kh); exact when C%32==0
    int cb = 0, ckw = 0, ckh = 0;

    for (int k0 = 0; k0 < K; k0 += 32) {
        // ---------------- stage A tile (128 x 32) ----------------
        if (amode == 0) {
            uint4 q0 = {0, 0, 0, 0}, q1 = {0, 0, 0, 0};
            if (am_ok) {
                const uint4* p = (const uint4*)(A + (size_t)am * K + (k0 + klo));
                q0 = p[0];
                q1 = p[1];
            }
            *(uint4*)&As[srow][klo]     = q0;
            *(uint4*)&As[srow][klo + 8] = q1;
        } else if (amode == 2) {
            const int ih = aihb + ckh;
            const int iw = aiwb + ckw;
            uint4 q0 = {0, 0, 0, 0}, q1 = {0, 0, 0, 0};
            if (am_ok && ih >= 0 && ih < H && iw >= 0 && iw < Wd) {
                const uint4* p = (const uint4*)
                    (A + ((size_t)((ab * H + ih) * Wd + iw)) * C + (cb + klo));
                q0 = p[0];
                q1 = p[1];
            }
            *(uint4*)&As[srow][klo]     = q0;
            *(uint4*)&As[srow][klo + 8] = q1;
        } else {
            // generic: one div per K step, incremental carry per element
            int kk = k0 + klo;
            int c  = kk % C;
            int t  = kk / C;
            int kw = t % KW;
            int kh = t / KW;
#pragma unroll
            for (int j = 0; j < 16; j++) {
                unsigned short v = 0;
                if (am_ok && (kk + j) < K) {
                    int ih = aihb + kh, iw = aiwb + kw;
                    if (ih >= 0 && ih < H && iw >= 0 && iw < Wd)
                        v = A[((size_t)((ab * H + ih) * Wd + iw)) * C + c];
                }
                As[srow][klo + j] = v;
                if (++c == C) { c = 0; if (++kw == KW) { kw = 0; ++kh; } }
            }
        }

        // ---------------- stage B tile (128 x 32), f32 -> bf16 ----------------
        if (bmode == 0) {
            const float* p = W + (size_t)bn * K + (k0 + klo);
#pragma unroll
            for (int j = 0; j < 8; j++) {
                unsigned v = 0;
                if (bn_ok) {
                    float f0 = p[2 * j];
                    float f1 = p[2 * j + 1];
                    v = (unsigned)f2bf(f0) | ((unsigned)f2bf(f1) << 16);
                }
                *(unsigned*)&Bs[srow][klo + 2 * j] = v;
            }
        } else if (bmode == 2) {
            const size_t khkw = (size_t)KH * KW;
            const float* p = W + (((size_t)bn * C + (cb + klo)) * KH + ckh) * KW + ckw;
#pragma unroll
            for (int j = 0; j < 8; j++) {
                unsigned v = 0;
                if (bn_ok) {
                    float f0 = p[(size_t)(2 * j) * khkw];
                    float f1 = p[(size_t)(2 * j + 1) * khkw];
                    v = (unsigned)f2bf(f0) | ((unsigned)f2bf(f1) << 16);
                }
                *(unsigned*)&Bs[srow][klo + 2 * j] = v;
            }
        } else {
            int kk = k0 + klo;
            int c  = kk % C;
            int t  = kk / C;
            int kw = t % KW;
            int kh = t / KW;
#pragma unroll
            for (int j = 0; j < 16; j++) {
                unsigned short v = 0;
                if (bn_ok && (kk + j) < K)
                    v = f2bf(W[((size_t)(bn * C + c) * KH + kh) * KW + kw]);
                Bs[srow][klo + j] = v;
                if (++c == C) { c = 0; if (++kw == KW) { kw = 0; ++kh; } }
            }
        }
        __syncthreads();

        // ---------------- fragments + 8 WMMAs ----------------
        FragU a[2], b[4];
#pragma unroll
        for (int i = 0; i < 2; i++) {
            const int arow = wrow * 32 + i * 16 + lr;
#pragma unroll
            for (int r = 0; r < 8; r++) {
                const int kp = 2 * r + ((r >= 4) ? 8 : 0) + 8 * hi;
                a[i].u[r] = *(const unsigned*)&As[arow][kp];
            }
        }
#pragma unroll
        for (int j = 0; j < 4; j++) {
            const int brow = wcol * 64 + j * 16 + lr;
#pragma unroll
            for (int r = 0; r < 8; r++) {
                const int kp = 2 * r + ((r >= 4) ? 8 : 0) + 8 * hi;
                b[j].u[r] = *(const unsigned*)&Bs[brow][kp];
            }
        }
#pragma unroll
        for (int i = 0; i < 2; i++)
#pragma unroll
            for (int j = 0; j < 4; j++)
                acc[i][j] = __builtin_amdgcn_wmma_f32_16x16x32_bf16(
                    false, a[i].v, false, b[j].v, (short)0, acc[i][j], false, false);
        __syncthreads();

        // advance (cbase, kw, kh) by 32 K positions (exact when C%32==0)
        cb += 32;
        if (cb >= C) { cb = 0; if (++ckw == KW) { ckw = 0; ++ckh; } }
    }

    // ---------------- epilogue: bias + relu + store ----------------
#pragma unroll
    for (int i = 0; i < 2; i++) {
#pragma unroll
        for (int j = 0; j < 4; j++) {
            const int n = blockN + wcol * 64 + j * 16 + lr;
#pragma unroll
            for (int v = 0; v < 8; v++) {
                const int m = blockM + wrow * 32 + i * 16 + 8 * hi + v;
                if (m < M && n < N) {
                    float val = acc[i][j][v];
                    if (bias) val += bias[n];
                    if (relu) val = val > 0.f ? val : 0.f;
                    if (outf32) outF[(size_t)m * N + n] = val;
                    else        outB[(size_t)m * N + n] = f2bf(val);
                }
            }
        }
    }
}

// ---------------------------------------------------------------------------
// Small helper kernels
// ---------------------------------------------------------------------------

// x: f32 NCHW [16,3,224,224] -> bf16 NHWC [16,224,224,3]
__global__ void to_nhwc_bf16_k(const float* __restrict__ x,
                               unsigned short* __restrict__ out) {
    const int Bn = 16, C = 3, H = 224;
    int idx = blockIdx.x * blockDim.x + threadIdx.x;
    const int total = Bn * H * H * C;
    if (idx >= total) return;
    int c = idx % C; int t = idx / C;
    int w = t % H;   t /= H;
    int h = t % H;   int b = t / H;
    out[idx] = f2bf(x[((size_t)(b * C + c) * H + h) * H + w]);
}

// maxpool 3x3 stride 2, bf16 NHWC
__global__ void pool_bf16_k(const unsigned short* __restrict__ in,
                            unsigned short* __restrict__ out,
                            int Bn, int H, int W, int C, int OH, int OW) {
    int idx = blockIdx.x * blockDim.x + threadIdx.x;
    const int total = Bn * OH * OW * C;
    if (idx >= total) return;
    int c = idx % C; int t = idx / C;
    int ow = t % OW; t /= OW;
    int oh = t % OH; int b = t / OH;
    float m = -3.4e38f;
    for (int di = 0; di < 3; di++) {
        int ih = 2 * oh + di;
        if (ih >= H) continue;
        for (int dj = 0; dj < 3; dj++) {
            int iw = 2 * ow + dj;
            if (iw >= W) continue;
            float v = bf2f(in[((size_t)(b * H + ih) * W + iw) * C + c]);
            m = fmaxf(m, v);
        }
    }
    out[idx] = f2bf(m);
}

// bf16 NHWC [16,6,6,256] -> bf16 [16,9216] in PyTorch C-major flatten order
__global__ void flatten_k(const unsigned short* __restrict__ in,
                          unsigned short* __restrict__ out) {
    int idx = blockIdx.x * blockDim.x + threadIdx.x;
    const int total = 16 * 9216;
    if (idx >= total) return;
    int b = idx / 9216;
    int r = idx % 9216;
    int c  = r / 36;
    int hw = r % 36;
    int h = hw / 6, w = hw % 6;
    out[idx] = in[((size_t)(b * 6 + h) * 6 + w) * 256 + c];
}

// gate conv: f32 NCHW input -> f32 NHWC [16,56,56,16], stride 4 pad 3, ReLU
__global__ void gate_conv_k(const float* __restrict__ x,
                            const float* __restrict__ w,
                            const float* __restrict__ bias,
                            float* __restrict__ out) {
    const int Bn = 16, C = 3, H = 224, OC = 16, OH = 56, KH = 7, S = 4, P = 3;
    int idx = blockIdx.x * blockDim.x + threadIdx.x;
    const int total = Bn * OH * OH * OC;
    if (idx >= total) return;
    int c = idx % OC; int t = idx / OC;
    int ow = t % OH;  t /= OH;
    int oh = t % OH;  int b = t / OH;
    float acc = bias[c];
    for (int ic = 0; ic < C; ic++)
        for (int kh = 0; kh < KH; kh++) {
            int ih = oh * S - P + kh;
            if (ih < 0 || ih >= H) continue;
            for (int kw = 0; kw < KH; kw++) {
                int iw = ow * S - P + kw;
                if (iw < 0 || iw >= H) continue;
                acc += x[((size_t)(b * C + ic) * H + ih) * H + iw] *
                       w[((size_t)(c * C + ic) * KH + kh) * KH + kw];
            }
        }
    out[idx] = acc > 0.f ? acc : 0.f;
}

// gate maxpool3s2: f32 NHWC [16,56,56,16] -> [16,27,27,16]
__global__ void gate_pool_k(const float* __restrict__ in,
                            float* __restrict__ out) {
    int idx = blockIdx.x * blockDim.x + threadIdx.x;
    const int total = 16 * 27 * 27 * 16;
    if (idx >= total) return;
    int c = idx % 16; int t = idx / 16;
    int ow = t % 27;  t /= 27;
    int oh = t % 27;  int b = t / 27;
    float m = -3.4e38f;
    for (int di = 0; di < 3; di++)
        for (int dj = 0; dj < 3; dj++) {
            int ih = 2 * oh + di, iw = 2 * ow + dj;
            m = fmaxf(m, in[((size_t)(b * 56 + ih) * 56 + iw) * 16 + c]);
        }
    out[idx] = m;
}

// gate adaptive avg pool 27->4 + PyTorch flatten -> f32 [16,256]
__global__ void gate_adapt_k(const float* __restrict__ in,
                             float* __restrict__ out) {
    int idx = blockIdx.x * blockDim.x + threadIdx.x;
    const int total = 16 * 16 * 4 * 4;
    if (idx >= total) return;
    int j = idx % 4; int t = idx / 4;
    int i = t % 4;   t /= 4;
    int c = t % 16;  int b = t / 16;
    int hs = (i * 27) / 4, he = ((i + 1) * 27 + 3) / 4;
    int ws = (j * 27) / 4, we = ((j + 1) * 27 + 3) / 4;
    float s = 0.f;
    for (int h = hs; h < he; h++)
        for (int w = ws; w < we; w++)
            s += in[((size_t)(b * 27 + h) * 27 + w) * 16 + c];
    out[b * 256 + c * 16 + i * 4 + j] = s / (float)((he - hs) * (we - ws));
}

// gate head: FC 256->64 (relu), FC 64->8, /clamp(T), top-2 softmax -> gates
__global__ void gate_head_k(const float* __restrict__ feat,
                            const float* __restrict__ w1, const float* __restrict__ b1,
                            const float* __restrict__ w2, const float* __restrict__ b2,
                            const float* __restrict__ temp,
                            float* __restrict__ gates) {
    int b = threadIdx.x;
    if (b >= 16) return;
    float h[64];
    for (int j = 0; j < 64; j++) {
        float a = b1[j];
        for (int k = 0; k < 256; k++) a += w1[j * 256 + k] * feat[b * 256 + k];
        h[j] = a > 0.f ? a : 0.f;
    }
    float T = temp[0];
    T = T < 0.5f ? 0.5f : (T > 5.f ? 5.f : T);
    float lg[8];
    for (int e = 0; e < 8; e++) {
        float a = b2[e];
        for (int j = 0; j < 64; j++) a += w2[e * 64 + j] * h[j];
        lg[e] = a / T;
    }
    int i1 = 0;
    for (int e = 1; e < 8; e++) if (lg[e] > lg[i1]) i1 = e;
    int i2 = -1;
    for (int e = 0; e < 8; e++) {
        if (e == i1) continue;
        if (i2 < 0 || lg[e] > lg[i2]) i2 = e;
    }
    float e2 = expf(lg[i2] - lg[i1]);
    float s  = 1.f + e2;
    float g1 = 1.f / s, g2 = e2 / s;
    float norm = g1 + g2 + 1e-10f;
    g1 /= norm; g2 /= norm;
    for (int e = 0; e < 8; e++) gates[b * 8 + e] = 0.f;
    gates[b * 8 + i1] = g1;
    gates[b * 8 + i2] = g2;
}

// out[b,d] = sum_e gates[b,e] * eo[e,b,d]
__global__ void combine_k(const float* __restrict__ gates,
                          const float* __restrict__ eo,
                          float* __restrict__ out) {
    int idx = blockIdx.x * blockDim.x + threadIdx.x;
    if (idx >= 160) return;
    int b = idx / 10, d = idx % 10;
    float s = 0.f;
    for (int e = 0; e < 8; e++)
        s += gates[b * 8 + e] * eo[((size_t)e * 16 + b) * 10 + d];
    out[idx] = s;
}

// ---------------------------------------------------------------------------
// Host side
// ---------------------------------------------------------------------------
static void launch_gemm(hipStream_t s,
                        const unsigned short* A, const float* W, const float* bias,
                        unsigned short* outB, float* outF,
                        int M, int N, int K, int amode,
                        int H, int Wd, int C, int OH, int OW,
                        int KH, int KW, int stride, int pad,
                        int bmode, int relu, int outf32) {
    dim3 grid((N + BT - 1) / BT, (M + BT - 1) / BT);
    wmma_gemm_k<<<grid, dim3(256), 0, s>>>(A, W, bias, outB, outF,
                                           M, N, K, amode, H, Wd, C, OH, OW,
                                           KH, KW, stride, pad, bmode, relu, outf32);
}

extern "C" void kernel_launch(void* const* d_in, const int* in_sizes, int n_in,
                              void* d_out, int out_size, void* d_ws, size_t ws_size,
                              hipStream_t stream) {
    (void)in_sizes; (void)n_in; (void)out_size; (void)ws_size;

    const float* x    = (const float*)d_in[0];
    const float* ecw1 = (const float*)d_in[1];  const float* ecb1 = (const float*)d_in[2];
    const float* ecw2 = (const float*)d_in[3];  const float* ecb2 = (const float*)d_in[4];
    const float* ecw3 = (const float*)d_in[5];  const float* ecb3 = (const float*)d_in[6];
    const float* ecw4 = (const float*)d_in[7];  const float* ecb4 = (const float*)d_in[8];
    const float* ecw5 = (const float*)d_in[9];  const float* ecb5 = (const float*)d_in[10];
    const float* efw1 = (const float*)d_in[11]; const float* efb1 = (const float*)d_in[12];
    const float* efw2 = (const float*)d_in[13]; const float* efb2 = (const float*)d_in[14];
    const float* efw3 = (const float*)d_in[15]; const float* efb3 = (const float*)d_in[16];
    const float* gcw  = (const float*)d_in[17]; const float* gcb  = (const float*)d_in[18];
    const float* gfw1 = (const float*)d_in[19]; const float* gfb1 = (const float*)d_in[20];
    const float* gfw2 = (const float*)d_in[21]; const float* gfb2 = (const float*)d_in[22];
    const float* temp = (const float*)d_in[23];
    float* out = (float*)d_out;

    // --- bump-allocate workspace ------------------------------------------
    char* ws = (char*)d_ws;
    size_t off = 0;
    auto alloc = [&](size_t bytes) -> char* {
        char* p = ws + off;
        off += (bytes + 255) & ~(size_t)255;
        return p;
    };
    unsigned short* XH   = (unsigned short*)alloc((size_t)16 * 224 * 224 * 3 * 2);
    unsigned short* BUF0 = (unsigned short*)alloc((size_t)3200000 * 2);
    unsigned short* BUF1 = (unsigned short*)alloc((size_t)3200000 * 2);
    float* GCONV = (float*)alloc((size_t)16 * 56 * 56 * 16 * 4);
    float* GPOOL = (float*)alloc((size_t)16 * 27 * 27 * 16 * 4);
    float* GFEAT = (float*)alloc((size_t)16 * 256 * 4);
    float* GATES = (float*)alloc((size_t)16 * 8 * 4);
    float* EOUT  = (float*)alloc((size_t)8 * 16 * 10 * 4);

    // --- input -> bf16 NHWC ------------------------------------------------
    {
        int total = 16 * 224 * 224 * 3;
        to_nhwc_bf16_k<<<(total + 255) / 256, 256, 0, stream>>>(x, XH);
    }

    // --- gate network ------------------------------------------------------
    {
        int t1 = 16 * 56 * 56 * 16;
        gate_conv_k<<<(t1 + 255) / 256, 256, 0, stream>>>(x, gcw, gcb, GCONV);
        int t2 = 16 * 27 * 27 * 16;
        gate_pool_k<<<(t2 + 255) / 256, 256, 0, stream>>>(GCONV, GPOOL);
        int t3 = 16 * 16 * 4 * 4;
        gate_adapt_k<<<(t3 + 255) / 256, 256, 0, stream>>>(GPOOL, GFEAT);
        gate_head_k<<<1, 32, 0, stream>>>(GFEAT, gfw1, gfb1, gfw2, gfb2, temp, GATES);
    }

    // --- experts -----------------------------------------------------------
    for (int e = 0; e < 8; e++) {
        // conv1: [16,224,224,3] -> [16,55,55,64], k11 s4 p2  (generic: C=3)
        launch_gemm(stream, XH, ecw1 + (size_t)e * 23232, ecb1 + (size_t)e * 64,
                    BUF0, nullptr,
                    16 * 55 * 55, 64, 3 * 11 * 11, /*amode*/1,
                    224, 224, 3, 55, 55, 11, 11, 4, 2,
                    /*bmode*/1, /*relu*/1, /*outf32*/0);
        {   // pool1: 55 -> 27
            int t = 16 * 27 * 27 * 64;
            pool_bf16_k<<<(t + 255) / 256, 256, 0, stream>>>(BUF0, BUF1, 16, 55, 55, 64, 27, 27);
        }
        // conv2: [16,27,27,64] -> [16,27,27,192], k5 s1 p2  (aligned: C=64)
        launch_gemm(stream, BUF1, ecw2 + (size_t)e * 307200, ecb2 + (size_t)e * 192,
                    BUF0, nullptr,
                    16 * 27 * 27, 192, 64 * 25, 2,
                    27, 27, 64, 27, 27, 5, 5, 1, 2, 2, 1, 0);
        {   // pool2: 27 -> 13
            int t = 16 * 13 * 13 * 192;
            pool_bf16_k<<<(t + 255) / 256, 256, 0, stream>>>(BUF0, BUF1, 16, 27, 27, 192, 13, 13);
        }
        // conv3: -> [16,13,13,384], k3 s1 p1  (aligned: C=192)
        launch_gemm(stream, BUF1, ecw3 + (size_t)e * 663552, ecb3 + (size_t)e * 384,
                    BUF0, nullptr,
                    16 * 13 * 13, 384, 192 * 9, 2,
                    13, 13, 192, 13, 13, 3, 3, 1, 1, 2, 1, 0);
        // conv4: -> [16,13,13,256]  (aligned: C=384)
        launch_gemm(stream, BUF0, ecw4 + (size_t)e * 884736, ecb4 + (size_t)e * 256,
                    BUF1, nullptr,
                    16 * 13 * 13, 256, 384 * 9, 2,
                    13, 13, 384, 13, 13, 3, 3, 1, 1, 2, 1, 0);
        // conv5: -> [16,13,13,256]  (aligned: C=256)
        launch_gemm(stream, BUF1, ecw5 + (size_t)e * 589824, ecb5 + (size_t)e * 256,
                    BUF0, nullptr,
                    16 * 13 * 13, 256, 256 * 9, 2,
                    13, 13, 256, 13, 13, 3, 3, 1, 1, 2, 1, 0);
        {   // pool5: 13 -> 6
            int t = 16 * 6 * 6 * 256;
            pool_bf16_k<<<(t + 255) / 256, 256, 0, stream>>>(BUF0, BUF1, 16, 13, 13, 256, 6, 6);
        }
        {   // flatten NHWC -> C-major [16, 9216]
            int t = 16 * 9216;
            flatten_k<<<(t + 255) / 256, 256, 0, stream>>>(BUF1, BUF0);
        }
        // fc1: [16,9216] x [9216,4096], relu
        launch_gemm(stream, BUF0, efw1 + (size_t)e * 37748736, efb1 + (size_t)e * 4096,
                    BUF1, nullptr,
                    16, 4096, 9216, 0,
                    0, 0, 1, 0, 0, 1, 1, 1, 0, 0, 1, 0);
        // fc2: [16,4096] x [4096,4096], relu
        launch_gemm(stream, BUF1, efw2 + (size_t)e * 16777216, efb2 + (size_t)e * 4096,
                    BUF0, nullptr,
                    16, 4096, 4096, 0,
                    0, 0, 1, 0, 0, 1, 1, 1, 0, 0, 1, 0);
        // fc3: [16,4096] x [4096,10] -> f32 expert output
        launch_gemm(stream, BUF0, efw3 + (size_t)e * 40960, efb3 + (size_t)e * 10,
                    nullptr, EOUT + (size_t)e * 160,
                    16, 10, 4096, 0,
                    0, 0, 1, 0, 0, 1, 1, 1, 0, 0, 0, 1);
    }

    // --- gated combine -----------------------------------------------------
    combine_k<<<1, 160, 0, stream>>>(GATES, EOUT, out);
}
